// TopKSAE_27530740367606
// MI455X (gfx1250) — matrix operand order repaired and checked by reading, not compiled, not run on hardware
//
#include <hip/hip_runtime.h>
#include <hip/hip_bf16.h>

#define DD 2048      // activation dim
#define FF 16384     // dict size
#define BB 8192      // batch
#define KK 64        // top-k

typedef __attribute__((ext_vector_type(16))) __bf16 bf16x16;
typedef __attribute__((ext_vector_type(8)))  __bf16 bf16x8;
typedef __attribute__((ext_vector_type(4)))  __bf16 bf16x4;
typedef __attribute__((ext_vector_type(8)))  float  f32x8;

// ---------------------------------------------------------------------------
// Kernel: convert W_enc [F,D] f32 -> bf16 (vectorized 4-wide)
// ---------------------------------------------------------------------------
__global__ void k_convert_wenc(const float* __restrict__ w, __bf16* __restrict__ o) {
    int i = blockIdx.x * 256 + threadIdx.x;          // group of 4 elements
    float4 v = ((const float4*)w)[i];
    bf16x4 r = { (__bf16)v.x, (__bf16)v.y, (__bf16)v.z, (__bf16)v.w };
    ((bf16x4*)o)[i] = r;
}

// ---------------------------------------------------------------------------
// Kernel: xm = (x - b_dec) -> bf16, [B,D]
// ---------------------------------------------------------------------------
__global__ void k_prep_x(const float* __restrict__ x, const float* __restrict__ bdec,
                         __bf16* __restrict__ o) {
    int i = blockIdx.x * 256 + threadIdx.x;          // group of 4 elements
    int d = (i * 4) & (DD - 1);                      // D is power of two
    float4 v  = ((const float4*)x)[i];
    float4 bd = *(const float4*)(bdec + d);
    bf16x4 r = { (__bf16)(v.x - bd.x), (__bf16)(v.y - bd.y),
                 (__bf16)(v.z - bd.z), (__bf16)(v.w - bd.w) };
    ((bf16x4*)o)[i] = r;
}

// ---------------------------------------------------------------------------
// Kernel: W_decT[f,d] = W_dec[d,f]   (tiled 32x32 transpose through LDS)
// ---------------------------------------------------------------------------
__global__ void k_transpose_wdec(const float* __restrict__ wd, float* __restrict__ wdt) {
    __shared__ float tile[32][33];
    const int f0 = blockIdx.x * 32, d0 = blockIdx.y * 32;
    const int tx = threadIdx.x, ty = threadIdx.y;    // block 32x8
#pragma unroll
    for (int i = 0; i < 32; i += 8)
        tile[ty + i][tx] = wd[(size_t)(d0 + ty + i) * FF + f0 + tx];
    __syncthreads();
#pragma unroll
    for (int i = 0; i < 32; i += 8)
        wdt[(size_t)(f0 + ty + i) * DD + d0 + tx] = tile[tx][ty + i];
}

// ---------------------------------------------------------------------------
// Fused encode (bf16 WMMA, f32 accumulate) + per-row top-64.
// Block = 256 threads (8 waves). Block tile = 32 rows x 128 cols per chunk.
// Wave w: M-tile (w&1), N-subrange (w>>1)*32 -> two 16x16 WMMA tiles sharing
// one A fragment (two accumulators). After staging pre-activations in LDS:
//   stage 1: all 8 waves ballot-filter candidates vs each row's heap-min,
//            publishing 32-bit masks to LDS (no atomics, no clears).
//   stage 2: 32 row-owner lanes insert only flagged candidates into per-row
//            64-entry min-heaps in LDS (expected ~3 inserts/row/chunk).
// ---------------------------------------------------------------------------
__global__ __launch_bounds__(256)
void k_encode_topk(const __bf16* __restrict__ xm, const __bf16* __restrict__ we,
                   const float* __restrict__ benc,
                   float* __restrict__ tkVal, int* __restrict__ tkIdx) {
    __shared__ float    preTile[32 * 128];   // 16 KB
    __shared__ float    heapVal[32 * 64];    // 8 KB
    __shared__ int      heapIdx[32 * 64];    // 8 KB
    __shared__ unsigned candMask[32 * 4];    // 512 B : 128 candidate bits per row

    const int tid   = threadIdx.x;
    const int wave  = tid >> 5;
    const int lane  = tid & 31;          // wave32
    const int half  = lane >> 4;
    const int mn    = lane & 15;
    const int mTile = wave & 1;          // 2 M-tiles of 16 rows
    const int nSub  = wave >> 1;         // 4 N-subranges of 32 cols
    const int rowBase = blockIdx.x * 32;

    for (int i = tid; i < 32 * 64; i += 256) { heapVal[i] = -3.0e38f; heapIdx[i] = 0; }
    __syncthreads();

    // A operand: row (rowBase + mTile*16 + mn); per K-chunk of 32 the lane loads
    // contiguous 16 bf16 at K-offset 16*half (same k order as B).
    const __bf16* aPtr = xm + (size_t)(rowBase + mTile * 16 + mn) * DD;

    for (int fc = 0; fc < FF / 128; ++fc) {
        const int colBase = fc * 128 + nSub * 32;
        // B operands: rows (colBase + mn) and (colBase + 16 + mn) of W_enc_bf16;
        // contiguous 16 bf16 at K-offset 16*half (same k order as A).
        const __bf16* bPtr0 = we + (size_t)(colBase + mn) * DD;
        const __bf16* bPtr1 = we + (size_t)(colBase + 16 + mn) * DD;

        f32x8 acc0 = {};
        f32x8 acc1 = {};
#pragma unroll 4
        for (int kc = 0; kc < DD / 32; ++kc) {
            const int kb = kc * 32;
            union { bf16x16 v; bf16x8 h[2]; } A;
            A.h[0] = *(const bf16x8*)(aPtr + kb + 16 * half);
            A.h[1] = *(const bf16x8*)(aPtr + kb + 16 * half + 8);
            bf16x16 B0 = *(const bf16x16*)(bPtr0 + kb + 16 * half);
            bf16x16 B1 = *(const bf16x16*)(bPtr1 + kb + 16 * half);
            acc0 = __builtin_amdgcn_wmma_f32_16x16x32_bf16(
                       false, A.v, false, B0, (short)0, acc0, false, false);
            acc1 = __builtin_amdgcn_wmma_f32_16x16x32_bf16(
                       false, A.v, false, B1, (short)0, acc1, false, false);
        }

        const float bias0 = benc[colBase + mn];
        const float bias1 = benc[colBase + 16 + mn];
#pragma unroll
        for (int r = 0; r < 8; ++r) {  // C layout: element r -> row (r + 8*half), col mn
            const int prow = (mTile * 16 + half * 8 + r) * 128 + nSub * 32 + mn;
            preTile[prow]      = acc0[r] + bias0;
            preTile[prow + 16] = acc1[r] + bias1;
        }
        __syncthreads();

        // ---- stage 1: wave-parallel candidate filter (ballot per 32 cands) ----
        {
#pragma unroll
            for (int r = 0; r < 4; ++r) {
                const int row = wave * 4 + r;
                const float rootMin = heapVal[row * 64];
#pragma unroll
                for (int j = 0; j < 4; ++j) {
                    const bool p = preTile[row * 128 + j * 32 + lane] > rootMin;
                    const unsigned m = __builtin_amdgcn_ballot_w32(p);
                    if (lane == j) candMask[row * 4 + j] = m;
                }
            }
        }
        __syncthreads();

        // ---- stage 2: per-row serial insertion of flagged candidates ----
        if (tid < 32) {
            const int row   = tid;
            const int hbase = row * 64;
            const int pbase = row * 128;
            const int gcol  = fc * 128;
#pragma unroll 1
            for (int w = 0; w < 4; ++w) {
                unsigned m = candMask[row * 4 + w];
                while (m) {
                    const int L = __builtin_ctz(m);
                    m &= m - 1;
                    const int c = w * 32 + L;
                    const float v = preTile[pbase + c];
                    if (v > heapVal[hbase]) {          // recheck vs current min
                        int i = 0;
                        for (;;) {
                            int l = 2 * i + 1;
                            if (l >= 64) break;
                            int r2 = l + 1;
                            int s = (r2 < 64 && heapVal[hbase + r2] < heapVal[hbase + l]) ? r2 : l;
                            if (heapVal[hbase + s] >= v) break;
                            heapVal[hbase + i] = heapVal[hbase + s];
                            heapIdx[hbase + i] = heapIdx[hbase + s];
                            i = s;
                        }
                        heapVal[hbase + i] = v;
                        heapIdx[hbase + i] = gcol + c;
                    }
                }
            }
        }
        __syncthreads();
    }

    for (int e = tid; e < 32 * 64; e += 256) {
        int m = e >> 6, j = e & 63;
        tkVal[(size_t)(rowBase + m) * KK + j] = heapVal[e];
        tkIdx[(size_t)(rowBase + m) * KK + j] = heapIdx[e];
    }
}

// ---------------------------------------------------------------------------
// Kernel: sparse decode. One block per token; 64 (val,idx) pairs in LDS.
// TR=true: wd is W_decT [F,D] -> contiguous coalesced float4 rows.
// TR=false: fallback, strided gathers from W_dec [D,F].
// ---------------------------------------------------------------------------
template <bool TR>
__global__ __launch_bounds__(256)
void k_decode(const float* __restrict__ tkVal, const int* __restrict__ tkIdx,
              const float* __restrict__ wd, const float* __restrict__ bdec,
              float* __restrict__ out) {
    __shared__ float sval[KK];
    __shared__ int   sidx[KK];
    const int b = blockIdx.x, tid = threadIdx.x;
    if (tid < KK) {
        sval[tid] = tkVal[(size_t)b * KK + tid];
        sidx[tid] = tkIdx[(size_t)b * KK + tid];
    }
    __syncthreads();

    if (TR) {
        float4 acc0 = *(const float4*)(bdec + tid * 4);
        float4 acc1 = *(const float4*)(bdec + 1024 + tid * 4);
        for (int k = 0; k < KK; ++k) {
            const float  v   = sval[k];
            const float* row = wd + (size_t)sidx[k] * DD;
            float4 w0 = *(const float4*)(row + tid * 4);
            float4 w1 = *(const float4*)(row + 1024 + tid * 4);
            acc0.x += v * w0.x; acc0.y += v * w0.y; acc0.z += v * w0.z; acc0.w += v * w0.w;
            acc1.x += v * w1.x; acc1.y += v * w1.y; acc1.z += v * w1.z; acc1.w += v * w1.w;
        }
        *(float4*)(out + (size_t)b * DD + tid * 4) = acc0;
        *(float4*)(out + (size_t)b * DD + 1024 + tid * 4) = acc1;
    } else {
        float acc[8];
#pragma unroll
        for (int j = 0; j < 8; ++j) acc[j] = bdec[tid + j * 256];
        for (int k = 0; k < KK; ++k) {
            const float v = sval[k];
            const int   f = sidx[k];
#pragma unroll
            for (int j = 0; j < 8; ++j)
                acc[j] += v * wd[(size_t)(tid + j * 256) * FF + f];
        }
#pragma unroll
        for (int j = 0; j < 8; ++j) out[(size_t)b * DD + tid + j * 256] = acc[j];
    }
}

// ---------------------------------------------------------------------------
extern "C" void kernel_launch(void* const* d_in, const int* in_sizes, int n_in,
                              void* d_out, int out_size, void* d_ws, size_t ws_size,
                              hipStream_t stream) {
    const float* x    = (const float*)d_in[0];   // [B,D]
    const float* Wenc = (const float*)d_in[1];   // [F,D]
    const float* benc = (const float*)d_in[2];   // [F]
    const float* Wdec = (const float*)d_in[3];   // [D,F]
    const float* bdec = (const float*)d_in[4];   // [D]
    float* out = (float*)d_out;                  // [B,D]

    char* ws = (char*)d_ws;
    const size_t offWe  = 0;                                  // F*D bf16 (64 MB)
    const size_t offXm  = offWe  + (size_t)FF * DD * 2;       // B*D bf16 (32 MB)
    const size_t offVal = offXm  + (size_t)BB * DD * 2;       // B*K f32
    const size_t offIdx = offVal + (size_t)BB * KK * 4;       // B*K i32
    const size_t offWdT = offIdx + (size_t)BB * KK * 4;       // F*D f32 (128 MB)
    const size_t needTr = offWdT + (size_t)FF * DD * 4;
    const bool useTr = (ws_size >= needTr);

    __bf16* we_bf = (__bf16*)(ws + offWe);
    __bf16* xm_bf = (__bf16*)(ws + offXm);
    float*  tkVal = (float*)(ws + offVal);
    int*    tkIdx = (int*)(ws + offIdx);
    float*  wdT   = (float*)(ws + offWdT);

    k_convert_wenc<<<(FF * DD / 4) / 256, 256, 0, stream>>>(Wenc, we_bf);
    k_prep_x<<<(BB * DD / 4) / 256, 256, 0, stream>>>(x, bdec, xm_bf);
    if (useTr)
        k_transpose_wdec<<<dim3(FF / 32, DD / 32), dim3(32, 8), 0, stream>>>(Wdec, wdT);

    k_encode_topk<<<BB / 32, 256, 0, stream>>>(xm_bf, we_bf, benc, tkVal, tkIdx);

    if (useTr) k_decode<true ><<<BB, 256, 0, stream>>>(tkVal, tkIdx, wdT,  bdec, out);
    else       k_decode<false><<<BB, 256, 0, stream>>>(tkVal, tkIdx, Wdec, bdec, out);
}
